// GCN_40802189312205
// MI455X (gfx1250) — compile-verified, matrix operand order checked by reading, not run on hardware
//
#include <hip/hip_runtime.h>
#include <hip/hip_bf16.h>

#define NNODES 100000
#define NEDGES 1600000
#define CH     128
#define OUTC   64
#define NGRAPH 64

typedef __attribute__((ext_vector_type(16))) __bf16 v16bf;
typedef __attribute__((ext_vector_type(8)))  float  v8f;

// ---------- helpers ----------
static __device__ __forceinline__ unsigned short f2bf(float f) {
    unsigned u = __float_as_uint(f);
    u += 0x7FFFu + ((u >> 16) & 1u);          // round-to-nearest-even
    return (unsigned short)(u >> 16);
}
static __device__ __forceinline__ unsigned pack2(float a, float b) {
    return (unsigned)f2bf(a) | ((unsigned)f2bf(b) << 16);
}

// ---------- kernels ----------
__global__ void k_fill(float* __restrict__ p, float v, int n) {
    int i = blockIdx.x * blockDim.x + threadIdx.x;
    if (i < n) p[i] = v;
}

// deg[dst] += 1 per edge (deg pre-filled with 1.0 for the self loop)
__global__ void k_deg(const long long* __restrict__ ei, float* __restrict__ deg) {
    int e = blockIdx.x * blockDim.x + threadIdx.x;
    if (e < NEDGES) unsafeAtomicAdd(deg + (int)ei[NEDGES + e], 1.0f);
}

__global__ void k_dinv(const float* __restrict__ deg, float* __restrict__ dinv) {
    int i = blockIdx.x * blockDim.x + threadIdx.x;
    if (i < NNODES) dinv[i] = rsqrtf(deg[i]);
}

// fp32 [rows x 128] -> packed bf16 pairs (dword = 2 consecutive K)
__global__ void k_cvt(const float* __restrict__ src, unsigned* __restrict__ dst, int nPairs) {
    int i = blockIdx.x * blockDim.x + threadIdx.x;
    if (i < nPairs) dst[i] = pack2(src[2 * i], src[2 * i + 1]);
}

// W[k*128+n] (fp32, K-major) -> Wt packed bf16: wt[n*64 + kp] = {W[2kp][n], W[2kp+1][n]}
__global__ void k_cvtWT(const float* __restrict__ W, unsigned* __restrict__ wt) {
    int i = blockIdx.x * blockDim.x + threadIdx.x;   // 128*64
    if (i >= CH * 64) return;
    int n = i >> 6, kp = i & 63;
    wt[i] = pack2(W[(2 * kp) * CH + n], W[(2 * kp + 1) * CH + n]);
}

// C[100000x128] = A_bf16[100000x128] x Wt_bf16(N-major)[128x128]; fp32 accumulate.
// 8 waves / block, one 16x16 tile per wave, K in chunks of 32 -> 4 WMMAs.
__global__ void __launch_bounds__(256) k_gemm(const unsigned* __restrict__ A,
                                              const unsigned* __restrict__ Bt,
                                              float* __restrict__ Cf) {
    int wave = threadIdx.x >> 5;
    int tile = blockIdx.x * 8 + wave;        // 50000 tiles = 6250 m-tiles * 8 n-tiles
    int mt = tile >> 3, nt = tile & 7;
    int lane = threadIdx.x & 31;
    int half = lane >> 4, idx = lane & 15;

    const unsigned* arow = A  + (size_t)(mt * 16 + idx) * 64;   // 64 dwords = 128 bf16 per row
    const unsigned* brow = Bt + (size_t)(nt * 16 + idx) * 64;

    v8f c = {};
#pragma unroll
    for (int kk = 0; kk < CH; kk += 32) {
        // A frag 16x32 bf16: lanes 0-15 row M=idx hold K=kk..kk+7 (v0-3), kk+16..kk+23 (v4-7);
        // lanes 16-31 same rows, K shifted by +8.  Dword pairs are consecutive K -> direct loads.
        int base = (kk >> 1) + half * 4;
        union { unsigned u[8]; v16bf v; } a, b;
        *(uint4*)&a.u[0] = *(const uint4*)(arow + base);
        *(uint4*)&a.u[4] = *(const uint4*)(arow + base + 8);
        *(uint4*)&b.u[0] = *(const uint4*)(brow + base);
        *(uint4*)&b.u[4] = *(const uint4*)(brow + base + 8);
        c = __builtin_amdgcn_wmma_f32_16x16x32_bf16(false, a.v, false, b.v,
                                                    (short)0, c, false, false);
    }
    // D layout: VGPR r -> row = mt*16 + half*8 + r, col = nt*16 + idx
    float* out = Cf + (size_t)(mt * 16 + half * 8) * CH + nt * 16 + idx;
#pragma unroll
    for (int r = 0; r < 8; ++r) out[(size_t)r * CH] = c[r];
}

// one wave per edge: acc[dst] += dinv[src]*dinv[dst] * h[src]; float4 per lane
__global__ void k_scatter(const long long* __restrict__ ei,
                          const float* __restrict__ h,
                          const float* __restrict__ dinv,
                          float* __restrict__ acc) {
    int t = blockIdx.x * blockDim.x + threadIdx.x;
    int e = t >> 5;
    if (e >= NEDGES) return;
    int lane = threadIdx.x & 31;
    int s = (int)ei[e];
    int d = (int)ei[NEDGES + e];
    float nrm = dinv[s] * dinv[d];
    float4 v = ((const float4*)(h + (size_t)s * CH))[lane];
    float* o = acc + (size_t)d * CH + lane * 4;
    unsafeAtomicAdd(o + 0, nrm * v.x);
    unsafeAtomicAdd(o + 1, nrm * v.y);
    unsafeAtomicAdd(o + 2, nrm * v.z);
    unsafeAtomicAdd(o + 3, nrm * v.w);
}

// h = relu(acc + dinv^2 * h + bias); also emit packed bf16 for next layer's GEMM
__global__ void k_epilogue(float* __restrict__ h,
                           const float* __restrict__ acc,
                           const float* __restrict__ dinv,
                           const float* __restrict__ bias,
                           unsigned* __restrict__ hbf) {
    int t = blockIdx.x * blockDim.x + threadIdx.x;   // NNODES*64
    if (t >= NNODES * 64) return;
    int i = t >> 6, p = t & 63;
    float sl = dinv[i] * dinv[i];
    int b0 = i * CH + 2 * p;
    float a0 = fmaxf(acc[b0]     + sl * h[b0]     + bias[2 * p],     0.0f);
    float a1 = fmaxf(acc[b0 + 1] + sl * h[b0 + 1] + bias[2 * p + 1], 0.0f);
    h[b0] = a0; h[b0 + 1] = a1;
    hbf[t] = pack2(a0, a1);
}

// per-graph sums + counts (accumulator is 32KB -> L2 resident)
__global__ void k_pool(const float* __restrict__ h,
                       const long long* __restrict__ batch,
                       float* __restrict__ pooled, float* __restrict__ cnt) {
    int t = blockIdx.x * blockDim.x + threadIdx.x;   // NNODES*32
    int i = t >> 5;
    if (i >= NNODES) return;
    int q = t & 31;
    int g = (int)batch[i];
    float4 v = ((const float4*)(h + (size_t)i * CH))[q];
    float* o = pooled + g * CH + q * 4;
    unsafeAtomicAdd(o + 0, v.x);
    unsafeAtomicAdd(o + 1, v.y);
    unsafeAtomicAdd(o + 2, v.z);
    unsafeAtomicAdd(o + 3, v.w);
    if (q == 0) unsafeAtomicAdd(cnt + g, 1.0f);
}

// out[g][o] = (pooled[g]/cnt[g]) @ Wlin + blin   (mean applied after dot: linear)
__global__ void k_final(const float* __restrict__ pooled, const float* __restrict__ cnt,
                        const float* __restrict__ Wlin, const float* __restrict__ blin,
                        float* __restrict__ out) {
    int t = blockIdx.x * blockDim.x + threadIdx.x;   // 64*64
    if (t >= NGRAPH * OUTC) return;
    int g = t >> 6, o = t & 63;
    float s = 0.0f;
#pragma unroll 4
    for (int c = 0; c < CH; ++c) s += pooled[g * CH + c] * Wlin[c * OUTC + o];
    out[t] = s / fmaxf(cnt[g], 1.0f) + blin[o];
}

// ---------- launcher ----------
extern "C" void kernel_launch(void* const* d_in, const int* in_sizes, int n_in,
                              void* d_out, int out_size, void* d_ws, size_t ws_size,
                              hipStream_t stream) {
    const float*     x     = (const float*)d_in[0];
    const long long* ei    = (const long long*)d_in[1];
    const long long* batch = (const long long*)d_in[2];
    const float* W[3] = { (const float*)d_in[3], (const float*)d_in[5], (const float*)d_in[7] };
    const float* B[3] = { (const float*)d_in[4], (const float*)d_in[6], (const float*)d_in[8] };
    const float* Wlin = (const float*)d_in[9];
    const float* blin = (const float*)d_in[10];
    float* out = (float*)d_out;

    // carve workspace (256B aligned)
    auto au = [](size_t v) { return (v + 255) & ~(size_t)255; };
    char* w = (char*)d_ws;
    float*    deg    = (float*)w;    w += au((size_t)NNODES * 4);
    float*    dinv   = (float*)w;    w += au((size_t)NNODES * 4);
    unsigned* actbf  = (unsigned*)w; w += au((size_t)NNODES * 64 * 4);   // packed bf16 acts
    float*    hA     = (float*)w;    w += au((size_t)NNODES * CH * 4);   // GEMM out / layer out
    float*    hB     = (float*)w;    w += au((size_t)NNODES * CH * 4);   // scatter accumulator
    unsigned* wt     = (unsigned*)w; w += au((size_t)CH * 64 * 4);       // packed bf16 W^T
    float*    pooled = (float*)w;    w += au((size_t)NGRAPH * CH * 4);
    float*    cnt    = (float*)w;    w += au((size_t)NGRAPH * 4);

    const int T = 256;
    auto nb = [](long long n, int t) { return (unsigned)((n + t - 1) / t); };

    // degrees (self loop = init 1.0) -> dinv
    k_fill<<<nb(NNODES, T), T, 0, stream>>>(deg, 1.0f, NNODES);
    k_deg<<<nb(NEDGES, T), T, 0, stream>>>(ei, deg);
    k_dinv<<<nb(NNODES, T), T, 0, stream>>>(deg, dinv);

    // bf16 input activations for layer 1
    k_cvt<<<nb((long long)NNODES * 64, T), T, 0, stream>>>(x, actbf, NNODES * 64);

    for (int l = 0; l < 3; ++l) {
        k_cvtWT<<<nb(CH * 64, T), T, 0, stream>>>(W[l], wt);
        k_gemm<<<6250, T, 0, stream>>>(actbf, wt, hA);
        k_fill<<<nb((long long)NNODES * CH, T), T, 0, stream>>>(hB, 0.0f, NNODES * CH);
        k_scatter<<<nb((long long)NEDGES * 32, T), T, 0, stream>>>(ei, hA, dinv, hB);
        k_epilogue<<<nb((long long)NNODES * 64, T), T, 0, stream>>>(hA, hB, dinv, B[l], actbf);
    }

    // mean pool + final linear
    k_fill<<<nb(NGRAPH * CH, T), T, 0, stream>>>(pooled, 0.0f, NGRAPH * CH);
    k_fill<<<1, 64, 0, stream>>>(cnt, 0.0f, NGRAPH);
    k_pool<<<nb((long long)NNODES * 32, T), T, 0, stream>>>(hA, batch, pooled, cnt);
    k_final<<<nb(NGRAPH * OUTC, T), T, 0, stream>>>(pooled, cnt, Wlin, blin, out);
}
